// DIFFormer_73572789781370
// MI455X (gfx1250) — compile-verified, matrix operand order checked by reading, not run on hardware
//
// DIFFormer forward for MI455X (gfx1250), wave32 + WMMA bf16 + async-LDS staging.
//
// Input index assumption (insertion-order flattening of setup_inputs):
//  0:x 1:edge_index 2:batch 3:fc0.W 4:fc0.b
//  5..16: bns[0..2]{g,b,rm,rv}
//  17..28: attn[0..1]{Wq,bq,Wk,bk,Wv,bv}
//  29..32: gcn[0..1]{W,b}
//  33:fc_out.W 34:fc_out.b
#include <hip/hip_runtime.h>
#include <hip/hip_bf16.h>

typedef __attribute__((ext_vector_type(16))) __bf16 v16bf;
typedef __attribute__((ext_vector_type(8)))  float  v8f;
typedef __attribute__((ext_vector_type(4)))  int    v4i;

#define AS_STRIDE 40   // LDS row stride (bf16 elems); 80B rows keep 16B alignment
#define TILE_ELEMS (128 * AS_STRIDE)
#define BN_EPS 1e-5f

// ---- CDNA5 async global->LDS copy (ASYNCcnt path), with safe fallback ----
#if defined(__has_builtin)
#  if __has_builtin(__builtin_amdgcn_global_load_async_to_lds_b128)
#    define HAVE_ASYNC_LDS 1
#  endif
#endif
#ifndef HAVE_ASYNC_LDS
#  define HAVE_ASYNC_LDS 0
#endif

typedef __attribute__((address_space(1))) v4i gas_v4i;
typedef __attribute__((address_space(3))) v4i las_v4i;

__device__ __forceinline__ void async_copy16(const void* gsrc, void* ldst) {
#if HAVE_ASYNC_LDS
  __builtin_amdgcn_global_load_async_to_lds_b128((gas_v4i*)gsrc, (las_v4i*)ldst, 0, 0);
#else
  *(uint4*)ldst = *(const uint4*)gsrc;
#endif
}

__device__ __forceinline__ void async_wait0() {
#if HAVE_ASYNC_LDS
#  if defined(__has_builtin) && __has_builtin(__builtin_amdgcn_s_wait_asynccnt)
  __builtin_amdgcn_s_wait_asynccnt(0);
#  else
  asm volatile("s_wait_asynccnt 0" ::: "memory");
#  endif
#endif
}

__device__ __forceinline__ __bf16 f2bf(float f) {
  union { float f; unsigned u; } a; a.f = f;
  unsigned r = (a.u + 0x7FFFu + ((a.u >> 16) & 1u)) >> 16;
  union { unsigned short s; __bf16 b; } o; o.s = (unsigned short)r;
  return o.b;
}

// Load one 16x16(x32) fragment from LDS per ISA 7.12.2 16-bit A layout:
// lane<16: K = 0..7 and 16..23 ; lane>=16: K = 8..15 and 24..31 (row = lane&15)
__device__ __forceinline__ v16bf load_frag(const __bf16* tile, int lm, int kh) {
  union { v16bf v; uint4 u[2]; } f;
  const __bf16* p = tile + lm * AS_STRIDE + kh * 8;
  f.u[0] = *(const uint4*)(p);
  f.u[1] = *(const uint4*)(p + 16);
  return f.v;
}

// ---------------------------------------------------------------------------
// Generic WMMA GEMM: C[M,Ncols] = act( A[M,K](f32) * B[K,Ncols] + bias )
// B supplied pre-transposed bf16: Bt[n, k] (rows padded to 128-multiple).
// Double-buffered LDS; A staged via f32->bf16 convert, B via async-LDS DMA.
// Block = 256 thr (8 waves as 4x2); block tile 128x128; wave tile 32x64.
// ---------------------------------------------------------------------------
__global__ __launch_bounds__(256)
void gemm_wmma_bf16(const float* __restrict__ A, int lda,
                    const __bf16* __restrict__ Bt, int ldb,
                    const float* __restrict__ bias,
                    const float* __restrict__ bns, const float* __restrict__ bnt,
                    int relu,
                    float* __restrict__ C, int ldc,
                    int M, int K, int Ncols)
{
  __shared__ __bf16 As[2][TILE_ELEMS];
  __shared__ __bf16 Bs[2][TILE_ELEMS];
  const int tid  = threadIdx.x;
  const int lane = tid & 31;
  const int wave = tid >> 5;
  const int wm = wave & 3;       // wave row  (0..3) -> 32 rows each
  const int wn = wave >> 2;      // wave col  (0..1) -> 64 cols each
  const int lm = lane & 15;
  const int kh = lane >> 4;
  const int m0 = blockIdx.x * 128;
  const int n0 = blockIdx.y * 128;

  v8f acc[2][4];
#pragma unroll
  for (int i = 0; i < 2; i++)
#pragma unroll
    for (int j = 0; j < 4; j++)
#pragma unroll
      for (int e = 0; e < 8; e++) acc[i][j][e] = 0.0f;

  const int r  = tid >> 1;          // 0..127 : tile row staged by this thread
  const int kk = (tid & 1) * 16;    // 0 or 16: K segment
  const int gm = m0 + r;
  const bool arow_ok = (gm < M);
  const float*  aptr = A  + (size_t)gm * lda + kk;
  const __bf16* bptr = Bt + (size_t)(n0 + r) * ldb + kk;

  float areg[16];
  auto loadA = [&](int k0) {
    if (arow_ok) {
      const float4* s = (const float4*)(aptr + k0);
      *(float4*)&areg[0]  = s[0];
      *(float4*)&areg[4]  = s[1];
      *(float4*)&areg[8]  = s[2];
      *(float4*)&areg[12] = s[3];
    } else {
#pragma unroll
      for (int j = 0; j < 16; j++) areg[j] = 0.0f;
    }
  };
  auto storeA = [&](int buf) {
    union { __bf16 b[16]; uint4 u[2]; } tb;
#pragma unroll
    for (int j = 0; j < 16; j++) tb.b[j] = f2bf(areg[j]);
    uint4* dA = (uint4*)&As[buf][r * AS_STRIDE + kk];
    dA[0] = tb.u[0]; dA[1] = tb.u[1];
  };
  auto stageB = [&](int k0, int buf) {   // 16 bf16 = two b128 async copies
    __bf16* d = &Bs[buf][r * AS_STRIDE + kk];
    async_copy16(bptr + k0,     d);
    async_copy16(bptr + k0 + 8, d + 8);
  };

  loadA(0);
  stageB(0, 0);
  int buf = 0;
  for (int k0 = 0; k0 < K; k0 += 32) {
    storeA(buf);
    async_wait0();            // B tile (this buf) landed in LDS
    __syncthreads();          // A tile visible to all waves
    const int kn = k0 + 32;
    if (kn < K) {             // prefetch next tile while computing this one
      loadA(kn);
      stageB(kn, buf ^ 1);
    }

    v16bf afrag[2], bfrag[4];
#pragma unroll
    for (int mi = 0; mi < 2; mi++)
      afrag[mi] = load_frag(&As[buf][(wm * 32 + mi * 16) * AS_STRIDE], lm, kh);
#pragma unroll
    for (int ni = 0; ni < 4; ni++)
      bfrag[ni] = load_frag(&Bs[buf][(wn * 64 + ni * 16) * AS_STRIDE], lm, kh);
#pragma unroll
    for (int mi = 0; mi < 2; mi++)
#pragma unroll
      for (int ni = 0; ni < 4; ni++)
        acc[mi][ni] = __builtin_amdgcn_wmma_f32_16x16x32_bf16(
            false, afrag[mi], false, bfrag[ni], (short)0, acc[mi][ni], false, false);
    buf ^= 1;
  }

  // epilogue: C/D layout -> row = base + rr + (lane>=16 ? 8 : 0), col = base + (lane&15)
#pragma unroll
  for (int ni = 0; ni < 4; ni++) {
    const int col = n0 + wn * 64 + ni * 16 + lm;
    if (col >= Ncols) continue;
    const float bv = bias ? bias[col] : 0.0f;
    const float s  = bns  ? bns[col]  : 1.0f;
    const float tt = bnt  ? bnt[col]  : 0.0f;
#pragma unroll
    for (int mi = 0; mi < 2; mi++) {
      const int rowbase = m0 + wm * 32 + mi * 16 + kh * 8;
#pragma unroll
      for (int rr = 0; rr < 8; rr++) {
        const int row = rowbase + rr;
        if (row < M) {
          float vv = acc[mi][ni][rr] + bv;
          vv = vv * s + tt;
          if (relu) vv = fmaxf(vv, 0.0f);
          C[(size_t)row * ldc + col] = vv;
        }
      }
    }
  }
}

// ---------------------------------------------------------------------------
// KV-sum GEMM: kvs[h] += k_h^T * v_h over a node range (split-K, f32 atomics).
// Operands gathered transposed into LDS (K-dim = nodes); k/v live in fused
// qkv buffer with row stride 1536 (k at +512, v at +1024).
// grid = (splits, HEADS)
// ---------------------------------------------------------------------------
__global__ __launch_bounds__(256)
void kvs_wmma(const float* __restrict__ qkv, float* __restrict__ kvs, int N, int chunk)
{
  __shared__ __bf16 As[TILE_ELEMS];
  __shared__ __bf16 Bs[TILE_ELEMS];
  const int tid  = threadIdx.x;
  const int lane = tid & 31;
  const int wave = tid >> 5;
  const int wm = wave & 3, wn = wave >> 2;
  const int lm = lane & 15, kh = lane >> 4;
  const int h  = blockIdx.y;
  const int hb = h * 128;
  const int nstart = blockIdx.x * chunk;
  int nend = nstart + chunk; if (nend > N) nend = N;

  v8f acc[2][4];
#pragma unroll
  for (int i = 0; i < 2; i++)
#pragma unroll
    for (int j = 0; j < 4; j++)
#pragma unroll
      for (int e = 0; e < 8; e++) acc[i][j][e] = 0.0f;

  const int nn = tid >> 3;        // node-row within 32-step
  const int ms = (tid & 7) * 16;  // feature segment

  for (int n0 = nstart; n0 < nend; n0 += 32) {
    float ak[16], av[16];
    const int gn = n0 + nn;
    if (gn < nend) {
      const float4* ks = (const float4*)(qkv + (size_t)gn * 1536 + 512  + hb + ms);
      const float4* vs = (const float4*)(qkv + (size_t)gn * 1536 + 1024 + hb + ms);
      *(float4*)&ak[0] = ks[0]; *(float4*)&ak[4] = ks[1];
      *(float4*)&ak[8] = ks[2]; *(float4*)&ak[12] = ks[3];
      *(float4*)&av[0] = vs[0]; *(float4*)&av[4] = vs[1];
      *(float4*)&av[8] = vs[2]; *(float4*)&av[12] = vs[3];
    } else {
#pragma unroll
      for (int j = 0; j < 16; j++) { ak[j] = 0.0f; av[j] = 0.0f; }
    }
#pragma unroll
    for (int j = 0; j < 16; j++) {
      As[(ms + j) * AS_STRIDE + nn] = f2bf(ak[j]);   // A[m][n] = k[n, m]
      Bs[(ms + j) * AS_STRIDE + nn] = f2bf(av[j]);   // Bt[d][n] = v[n, d]
    }
    __syncthreads();

    v16bf afrag[2], bfrag[4];
#pragma unroll
    for (int mi = 0; mi < 2; mi++)
      afrag[mi] = load_frag(&As[(wm * 32 + mi * 16) * AS_STRIDE], lm, kh);
#pragma unroll
    for (int ni = 0; ni < 4; ni++)
      bfrag[ni] = load_frag(&Bs[(wn * 64 + ni * 16) * AS_STRIDE], lm, kh);
#pragma unroll
    for (int mi = 0; mi < 2; mi++)
#pragma unroll
      for (int ni = 0; ni < 4; ni++)
        acc[mi][ni] = __builtin_amdgcn_wmma_f32_16x16x32_bf16(
            false, afrag[mi], false, bfrag[ni], (short)0, acc[mi][ni], false, false);
    __syncthreads();
  }

  float* out = kvs + (size_t)h * 128 * 128;
#pragma unroll
  for (int ni = 0; ni < 4; ni++) {
    const int col = wn * 64 + ni * 16 + lm;
#pragma unroll
    for (int mi = 0; mi < 2; mi++) {
      const int rowbase = wm * 32 + mi * 16 + kh * 8;
#pragma unroll
      for (int rr = 0; rr < 8; rr++)
        atomicAdd(&out[(rowbase + rr) * 128 + col], acc[mi][ni][rr]);
    }
  }
}

// ---------------------------------------------------------------------------
// Small / elementwise kernels
// ---------------------------------------------------------------------------
__global__ void fill_f32(float* p, float v, long long n) {
  for (long long i = (long long)blockIdx.x * blockDim.x + threadIdx.x; i < n;
       i += (long long)gridDim.x * blockDim.x) p[i] = v;
}

__global__ void deg_accum(const int* __restrict__ dst, float* deg, int E) {
  int e = blockIdx.x * blockDim.x + threadIdx.x;
  if (e < E) atomicAdd(&deg[dst[e]], 1.0f);
}

__global__ void rsqrt_inplace(float* p, int n) {
  int i = blockIdx.x * blockDim.x + threadIdx.x;
  if (i < n) p[i] = rsqrtf(p[i]);
}

// W [K, Ncols] f32  ->  Bt [Npad, K] bf16 (zero-padded rows)
__global__ void convert_transpose(const float* __restrict__ W, __bf16* __restrict__ Bt,
                                  int K, int Ncols, int Npad) {
  const long long total = (long long)Npad * K;
  for (long long i = (long long)blockIdx.x * blockDim.x + threadIdx.x; i < total;
       i += (long long)gridDim.x * blockDim.x) {
    const int nrow = (int)(i / K), kk = (int)(i % K);
    const float v = (nrow < Ncols) ? W[(size_t)kk * Ncols + nrow] : 0.0f;
    Bt[i] = f2bf(v);
  }
}

__global__ void concat3(const float* __restrict__ a, const float* __restrict__ b,
                        const float* __restrict__ c, float* o, int seg) {
  int i = blockIdx.x * blockDim.x + threadIdx.x;
  if (i < seg) { o[i] = a[i]; o[i + seg] = b[i]; o[i + 2 * seg] = c[i]; }
}

__global__ void bn_prep(const float* g, const float* b, const float* rm, const float* rv,
                        float* s, float* t, int C) {
  int c = blockIdx.x * blockDim.x + threadIdx.x;
  if (c < C) { float sc = g[c] * rsqrtf(rv[c] + BN_EPS); s[c] = sc; t[c] = b[c] - rm[c] * sc; }
}

// L2-normalize the q and k head-rows of the fused qkv buffer.
// One wave per 128-wide subrow; subrows 0..7 of each node cover q(0..3),k(0..3).
__global__ __launch_bounds__(256) void normalize_qk(float* qkv, long long nrows) {
  const long long rr = (long long)blockIdx.x * 8 + (threadIdx.x >> 5);
  if (rr >= nrows) return;
  const int lane = threadIdx.x & 31;
  const long long n = rr >> 3; const int sub = (int)(rr & 7);
  float4* p = (float4*)(qkv + n * 1536 + sub * 128);
  float4 v = p[lane];
  float ss = v.x * v.x + v.y * v.y + v.z * v.z + v.w * v.w;
#pragma unroll
  for (int m = 16; m > 0; m >>= 1) ss += __shfl_xor(ss, m, 32);
  const float inv = rsqrtf(ss);
  v.x *= inv; v.y *= inv; v.z *= inv; v.w *= inv;
  p[lane] = v;
}

__global__ __launch_bounds__(256)
void colsum_kernel(const float* __restrict__ qkv, float* ksum, float* vsum, int N) {
  const int c = threadIdx.x;
  float a0 = 0, a1 = 0, b0 = 0, b1 = 0;
  for (int n = blockIdx.x; n < N; n += gridDim.x) {
    const float* row = qkv + (size_t)n * 1536;
    a0 += row[512 + c];  a1 += row[512 + c + 256];
    b0 += row[1024 + c]; b1 += row[1024 + c + 256];
  }
  atomicAdd(&ksum[c], a0); atomicAdd(&ksum[c + 256], a1);
  atomicAdd(&vsum[c], b0); atomicAdd(&vsum[c + 256], b1);
}

__global__ __launch_bounds__(256)
void denom_kernel(const float* __restrict__ qkv, const float* __restrict__ ksum,
                  float* denom, long long NH, float Nf) {
  const long long rr = (long long)blockIdx.x * 8 + (threadIdx.x >> 5);
  if (rr >= NH) return;
  const int lane = threadIdx.x & 31;
  const long long n = rr >> 2; const int h = (int)(rr & 3);
  const float4 qv = ((const float4*)(qkv + n * 1536 + h * 128))[lane];
  const float4 kv = ((const float4*)(ksum + h * 128))[lane];
  float d = qv.x * kv.x + qv.y * kv.y + qv.z * kv.z + qv.w * kv.w;
#pragma unroll
  for (int m = 16; m > 0; m >>= 1) d += __shfl_xor(d, m, 32);
  if (lane == 0) denom[rr] = d + Nf;
}

// x1[n,d] = 0.25 * sum_h num[n,h,d] / denom[n,h]; num lives in qkv k-slot (+512)
__global__ void x1_kernel(const float* __restrict__ qkv, const float* __restrict__ denom,
                          float* x1, long long total) {
  for (long long i = (long long)blockIdx.x * blockDim.x + threadIdx.x; i < total;
       i += (long long)gridDim.x * blockDim.x) {
    const long long n = i >> 7; const int d = (int)(i & 127);
    const float* num = qkv + n * 1536 + 512;
    float s = 0.0f;
#pragma unroll
    for (int h = 0; h < 4; h++) s += num[h * 128 + d] / denom[n * 4 + h];
    x1[i] = 0.25f * s;
  }
}

__global__ void gcn_init(const float* __restrict__ p, const float* __restrict__ dinv,
                         const float* __restrict__ bias, float* out, long long total) {
  for (long long i = (long long)blockIdx.x * blockDim.x + threadIdx.x; i < total;
       i += (long long)gridDim.x * blockDim.x) {
    const long long n = i >> 7; const int c = (int)(i & 127);
    const float di = dinv[n];
    out[i] = bias[c] + p[i] * di * di;   // self-loop: norm = dinv^2
  }
}

// One wave per edge, 32 lanes x float4 = 128 channels; f32 global atomics.
__global__ __launch_bounds__(256)
void gcn_scatter(const float* __restrict__ p, const int* __restrict__ src,
                 const int* __restrict__ dst, const float* __restrict__ dinv,
                 float* out, int E) {
  const int e = blockIdx.x * 8 + (threadIdx.x >> 5);
  if (e >= E) return;
  const int lane = threadIdx.x & 31;
  const int s = src[e], d = dst[e];
  const float nrm = dinv[s] * dinv[d];
  float* od = out + (size_t)d * 128;
  if (lane == 0) __builtin_prefetch(od, 1, 0);     // global_prefetch_b8
  const float4 v = ((const float4*)(p + (size_t)s * 128))[lane];
  float* o4 = od + lane * 4;
  atomicAdd(o4 + 0, v.x * nrm);
  atomicAdd(o4 + 1, v.y * nrm);
  atomicAdd(o4 + 2, v.z * nrm);
  atomicAdd(o4 + 3, v.w * nrm);
}

// h_out = relu(bn_affine(0.5*concat(x1,gcn) + 0.5*h_in))
__global__ void combine_kernel(const float* __restrict__ x1, const float* __restrict__ gout,
                               const float* __restrict__ hin, const float* __restrict__ s,
                               const float* __restrict__ t, float* hout, long long total) {
  for (long long i = (long long)blockIdx.x * blockDim.x + threadIdx.x; i < total;
       i += (long long)gridDim.x * blockDim.x) {
    const long long n = i >> 8; const int c = (int)(i & 255);
    const float z = (c < 128) ? x1[n * 128 + c] : gout[n * 128 + (c - 128)];
    float val = 0.5f * z + 0.5f * hin[i];
    val = val * s[c] + t[c];
    hout[i] = fmaxf(val, 0.0f);
  }
}

// ---------------------------------------------------------------------------
extern "C" void kernel_launch(void* const* d_in, const int* in_sizes, int n_in,
                              void* d_out, int out_size, void* d_ws, size_t ws_size,
                              hipStream_t stream)
{
  const int N = in_sizes[0] / 128;
  const int E = in_sizes[1] / 2;

  const float* x  = (const float*)d_in[0];
  const int*   ei = (const int*)d_in[1];
  const int* srcI = ei;
  const int* dstI = ei + E;

  const float* W0 = (const float*)d_in[3];
  const float* b0 = (const float*)d_in[4];
  const float *bng[3], *bnb[3], *bnrm[3], *bnrv[3];
  for (int j = 0; j < 3; j++) {
    bng[j]  = (const float*)d_in[5 + 4 * j];
    bnb[j]  = (const float*)d_in[6 + 4 * j];
    bnrm[j] = (const float*)d_in[7 + 4 * j];
    bnrv[j] = (const float*)d_in[8 + 4 * j];
  }
  const float *Wq[2], *bq[2], *Wk[2], *bk[2], *Wv[2], *bv[2];
  for (int i = 0; i < 2; i++) {
    const int b = 17 + 6 * i;
    Wq[i] = (const float*)d_in[b + 0]; bq[i] = (const float*)d_in[b + 1];
    Wk[i] = (const float*)d_in[b + 2]; bk[i] = (const float*)d_in[b + 3];
    Wv[i] = (const float*)d_in[b + 4]; bv[i] = (const float*)d_in[b + 5];
  }
  const float *Wg[2], *bg[2];
  for (int i = 0; i < 2; i++) {
    Wg[i] = (const float*)d_in[29 + 2 * i];
    bg[i] = (const float*)d_in[30 + 2 * i];
  }
  const float* Wo = (const float*)d_in[33];
  const float* bo = (const float*)d_in[34];

  // ---- workspace bump allocator ----
  char* base = (char*)d_ws;
  size_t off = 0;
  auto alloc = [&](size_t bytes) -> void* {
    void* p = base + off;
    off += (bytes + 255) & ~(size_t)255;
    return p;
  };
  float* h0   = (float*)alloc((size_t)N * 256 * 4);
  float* h1   = (float*)alloc((size_t)N * 256 * 4);
  float* qkv  = (float*)alloc((size_t)N * 1536 * 4);  // fused q|k|v ; k-slot reused as num
  float* pgcn = (float*)alloc((size_t)N * 128 * 4);
  float* gout = (float*)alloc((size_t)N * 128 * 4);
  float* x1b  = (float*)alloc((size_t)N * 128 * 4);
  float* dinv = (float*)alloc((size_t)N * 4);
  float* denb = (float*)alloc((size_t)N * 4 * 4);
  float* kvs  = (float*)alloc((size_t)4 * 128 * 128 * 4);
  float* ksum = (float*)alloc(512 * 4);
  float* vsum = (float*)alloc(512 * 4);
  float* bnS  = (float*)alloc(3 * 256 * 4);
  float* bnT  = (float*)alloc(3 * 256 * 4);
  float* bqkv[2];
  bqkv[0] = (float*)alloc(1536 * 4);
  bqkv[1] = (float*)alloc(1536 * 4);
  __bf16* W0t = (__bf16*)alloc((size_t)256 * 128 * 2);
  __bf16 *Wqkvt[2], *Wgt[2];
  for (int i = 0; i < 2; i++) {
    Wqkvt[i] = (__bf16*)alloc((size_t)1536 * 256 * 2);   // q|k|v stacked along N
    Wgt[i]   = (__bf16*)alloc((size_t)128 * 256 * 2);
  }
  __bf16* Wot   = (__bf16*)alloc((size_t)128 * 256 * 2);  // padded 40->128 rows
  __bf16* kvsBt = (__bf16*)alloc((size_t)4 * 128 * 128 * 2);
  (void)ws_size; (void)n_in; (void)out_size;

  const dim3 blk(256);
  const unsigned gM = (unsigned)((N + 127) / 128);

  // ---- one-time-per-call prep ----
  for (int j = 0; j < 3; j++)
    bn_prep<<<1, blk, 0, stream>>>(bng[j], bnb[j], bnrm[j], bnrv[j],
                                   bnS + j * 256, bnT + j * 256, 256);
  convert_transpose<<<512, blk, 0, stream>>>(W0, W0t, 128, 256, 256);
  for (int i = 0; i < 2; i++) {
    convert_transpose<<<512, blk, 0, stream>>>(Wq[i], Wqkvt[i],              256, 512, 512);
    convert_transpose<<<512, blk, 0, stream>>>(Wk[i], Wqkvt[i] + 512 * 256,  256, 512, 512);
    convert_transpose<<<512, blk, 0, stream>>>(Wv[i], Wqkvt[i] + 1024 * 256, 256, 512, 512);
    convert_transpose<<<512, blk, 0, stream>>>(Wg[i], Wgt[i], 256, 128, 128);
    concat3<<<2, blk, 0, stream>>>(bq[i], bk[i], bv[i], bqkv[i], 512);
  }
  convert_transpose<<<512, blk, 0, stream>>>(Wo, Wot, 256, 40, 128);

  // degree + dinv (self-loop => deg starts at 1)
  fill_f32<<<(N + 255) / 256, blk, 0, stream>>>(dinv, 1.0f, N);
  deg_accum<<<(E + 255) / 256, blk, 0, stream>>>(dstI, dinv, E);
  rsqrt_inplace<<<(N + 255) / 256, blk, 0, stream>>>(dinv, N);

  // ---- fc0: h0 = relu(bn0(x @ W0 + b0)) ----
  gemm_wmma_bf16<<<dim3(gM, 2), blk, 0, stream>>>(
      x, 128, W0t, 128, b0, bnS, bnT, 1, h0, 256, N, 128, 256);

  const int SPLITS = 96;
  const int chunk = (N + SPLITS - 1) / SPLITS;
  const long long NH = (long long)N * 4;

  for (int i = 0; i < 2; i++) {
    float* hin  = (i == 0) ? h0 : h1;
    float* hout = (i == 0) ? h1 : h0;

    // fused QKV projection: qkv = hin @ [Wq|Wk|Wv] + [bq|bk|bv]  (A read once)
    gemm_wmma_bf16<<<dim3(gM, 12), blk, 0, stream>>>(
        hin, 256, Wqkvt[i], 256, bqkv[i], nullptr, nullptr, 0, qkv, 1536, N, 256, 1536);

    // per-head L2 normalization of q and k (v untouched)
    normalize_qk<<<(unsigned)((N * 8LL + 7) / 8), blk, 0, stream>>>(qkv, (long long)N * 8);

    // kvs = sum_n k^T v  (split-K WMMA + f32 atomics), ksum, vsum
    fill_f32<<<256, blk, 0, stream>>>(kvs, 0.0f, 4 * 128 * 128);
    fill_f32<<<1, blk, 0, stream>>>(ksum, 0.0f, 512);
    fill_f32<<<1, blk, 0, stream>>>(vsum, 0.0f, 512);
    kvs_wmma<<<dim3(SPLITS, 4), blk, 0, stream>>>(qkv, kvs, N, chunk);
    colsum_kernel<<<512, blk, 0, stream>>>(qkv, ksum, vsum, N);

    // kvs -> transposed bf16 B operand per head
    for (int h = 0; h < 4; h++)
      convert_transpose<<<64, blk, 0, stream>>>(kvs + h * 16384, kvsBt + h * 16384,
                                                128, 128, 128);

    // num = q @ kvs + vsum (per head), written into the dead k-slot of qkv
    for (int h = 0; h < 4; h++)
      gemm_wmma_bf16<<<dim3(gM, 1), blk, 0, stream>>>(
          qkv + h * 128, 1536, kvsBt + h * 16384, 128, vsum + h * 128,
          nullptr, nullptr, 0, qkv + 512 + h * 128, 1536, N, 128, 128);

    // denom, x1 (mean over heads of num/denom)
    denom_kernel<<<(unsigned)((NH + 7) / 8), blk, 0, stream>>>(qkv, ksum, denb, NH, (float)N);
    x1_kernel<<<(unsigned)(((long long)N * 128 + 255) / 256), blk, 0, stream>>>(
        qkv, denb, x1b, (long long)N * 128);

    // GCN: p = hin @ Wg ; gout = b + D^-1/2 (A+I) D^-1/2 p
    gemm_wmma_bf16<<<dim3(gM, 1), blk, 0, stream>>>(
        hin, 256, Wgt[i], 256, nullptr, nullptr, nullptr, 0, pgcn, 128, N, 256, 128);
    gcn_init<<<(unsigned)(((long long)N * 128 + 255) / 256), blk, 0, stream>>>(
        pgcn, dinv, bg[i], gout, (long long)N * 128);
    gcn_scatter<<<(unsigned)((E + 7) / 8), blk, 0, stream>>>(pgcn, srcI, dstI, dinv, gout, E);

    // residual + BN(i+1) + relu
    combine_kernel<<<(unsigned)(((long long)N * 256 + 255) / 256), blk, 0, stream>>>(
        x1b, gout, hin, bnS + (i + 1) * 256, bnT + (i + 1) * 256, hout,
        (long long)N * 256);
  }

  // ---- fc_out: d_out = h @ Wo + bo  (h == h0 after layer 1) ----
  gemm_wmma_bf16<<<dim3(gM, 1), blk, 0, stream>>>(
      h0, 256, Wot, 256, bo, nullptr, nullptr, 0, (float*)d_out, 40, N, 256, 40);
}
